// PointerNetwork_1468878815487
// MI455X (gfx1250) — matrix operand order repaired and checked by reading, not compiled, not run on hardware
//
#include <hip/hip_runtime.h>
#include <hip/hip_bf16.h>
#include <math.h>

// ---------- types ----------
typedef __bf16 bf16;
typedef __attribute__((ext_vector_type(16))) __bf16 v16bf;
typedef __attribute__((ext_vector_type(8)))  __bf16 v8bf;
typedef __attribute__((ext_vector_type(8)))  float  v8f;
typedef __attribute__((ext_vector_type(8)))  float  v8ff;
typedef __attribute__((ext_vector_type(16))) float  v16f;
typedef __attribute__((ext_vector_type(4)))  float  v4f;
typedef __attribute__((ext_vector_type(4)))  int    v4i;

#define TGTN 2048
#define SRCN 2048
#define DIM  1024
#define NH   16
#define HD   64
#define VOC  32000
#define EVOC 2000
#define TOT  34000

// gfx1250 async load-to-LDS path (ASYNCcnt-tracked), with safe fallback
#if defined(__has_builtin)
#if __has_builtin(__builtin_amdgcn_global_load_async_to_lds_b128) && \
    __has_builtin(__builtin_amdgcn_s_wait_asynccnt)
#define HAVE_ASYNC_LDS 1
#endif
#endif
#ifndef HAVE_ASYNC_LDS
#define HAVE_ASYNC_LDS 0
#endif
#if HAVE_ASYNC_LDS
#define AS1 __attribute__((address_space(1)))
#define AS3 __attribute__((address_space(3)))
#endif

// packed f32x16 -> bf16x16 (lowers to v_cvt_pk_bf16_f32 x8)
__device__ __forceinline__ v16bf cvt16(v8ff lo, v8ff hi) {
  v16f t = __builtin_shufflevector(lo, hi, 0,1,2,3,4,5,6,7,8,9,10,11,12,13,14,15);
  return __builtin_convertvector(t, v16bf);
}

// bf16 A-fragment loader: elements 0..7 at p, 8..15 at p+16 (ISA A 16x32 layout)
__device__ __forceinline__ v16bf ld_a_frag(const bf16* p) {
  v8bf lo = *(const v8bf*)(p);
  v8bf hi = *(const v8bf*)(p + 16);
  return __builtin_shufflevector(lo, hi, 0,1,2,3,4,5,6,7,8,9,10,11,12,13,14,15);
}
// bf16 B-fragment loader: 16 contiguous K values per lane (contiguous k = s order, not the ISA table order)
__device__ __forceinline__ v16bf ld_b_frag(const bf16* p) {
  v8bf lo = *(const v8bf*)(p);
  v8bf hi = *(const v8bf*)(p + 8);
  return __builtin_shufflevector(lo, hi, 0,1,2,3,4,5,6,7,8,9,10,11,12,13,14,15);
}

// ---------------------------------------------------------------------------
// Kernel 1: C[2048,1024](bf16) = bf16( A[2048,1024] @ W[1024,1024]^T + bias )
// Block: 256 thr (8 waves). Block tile 16(M) x 128(N); wave tile 16x16 WMMA.
// A tile [16x32] staged to LDS once per WG via async-to-LDS, double-buffered.
// ---------------------------------------------------------------------------
__global__ void proj_gemm(const float* __restrict__ A, const float* __restrict__ W,
                          const float* __restrict__ bias, bf16* __restrict__ C) {
  __shared__ float As[2][16 * 32];
  const int tid  = threadIdx.x;
  const int lane = tid & 31, wave = tid >> 5;
  const int half = lane >> 4, l16 = lane & 15;
  const int rb  = blockIdx.x * 16;
  const int col = blockIdx.y * 128 + wave * 16 + l16;
  const float* Wr = W + (size_t)col * DIM;   // W[col, k] == (W^T)[k, col]

  // stage A[rb..rb+15, kb..kb+31] into As[buf]; 128 lanes x 16B
  auto stage = [&](int buf, int kb) {
    if (tid < 128) {
      const int row = tid >> 3, c4 = (tid & 7) << 2;
      const float* g = A + (size_t)(rb + row) * DIM + kb + c4;
      float* l = &As[buf][row * 32 + c4];
#if HAVE_ASYNC_LDS
      float* gnc = const_cast<float*>(g);
      __builtin_amdgcn_global_load_async_to_lds_b128((AS1 v4i*)gnc, (AS3 v4i*)l, 0, 0);
#else
      *(v4f*)l = *(const v4f*)g;
#endif
    }
  };

  stage(0, 0);
  v8f acc = {};
  int buf = 0;
  for (int kb = 0; kb < DIM; kb += 32) {
#if HAVE_ASYNC_LDS
    __builtin_amdgcn_s_wait_asynccnt(0);
#endif
    __syncthreads();                       // tile(buf) visible to all waves
    if (kb + 32 < DIM) stage(buf ^ 1, kb + 32);   // overlap next tile with WMMA

    const float* arow = &As[buf][l16 * 32] + half * 8;
    const v16bf a = cvt16(*(const v8ff*)(arow), *(const v8ff*)(arow + 16));
    const int kw = kb + half * 8;
    const v16bf b = cvt16(*(const v8ff*)(Wr + kw), *(const v8ff*)(Wr + kw + 16));
    acc = __builtin_amdgcn_wmma_f32_16x16x32_bf16(false, a, false, b,
                                                  (short)0, acc, false, false);
    __syncthreads();                       // all reads of As[buf] done before restage
    buf ^= 1;
  }

  const float bv = bias[col];
  #pragma unroll
  for (int r = 0; r < 8; ++r) {
    const int m = r + half * 8;                 // C layout: lane>=16 -> M+8
    C[(size_t)(rb + m) * DIM + col] = (bf16)(acc[r] + bv);
  }
}

// ---------------------------------------------------------------------------
// Kernel 2: per-head scores (WMMA, K=64) -> LDS softmax -> head-mean attn.
// One WG (8 waves) per 16 target rows. LDS: scores 128KB + accum 128KB.
// ---------------------------------------------------------------------------
__global__ void attn_kernel(const bf16* __restrict__ Q, const bf16* __restrict__ Kp,
                            float* __restrict__ attn) {
  extern __shared__ float lds[];
  float* sc   = lds;                 // [16][2048] scores for current head
  float* accb = lds + 16 * 2048;     // [16][2048] mean-over-heads accumulator
  float* red  = lds + 32 * 2048;     // [256] reduction scratch
  float* rowm = red + 256;           // [16]
  float* rowl = rowm + 16;           // [16]
  const int tid  = threadIdx.x;
  const int wave = tid >> 5, lane = tid & 31;
  const int half = lane >> 4, l16 = lane & 15;
  const int rb = blockIdx.x * 16;
  const int row = tid >> 4, sub = tid & 15;

  for (int i = tid; i < 16 * 2048; i += 256) accb[i] = 0.f;
  __syncthreads();

  for (int h = 0; h < NH; ++h) {
    // ---- scores tile via WMMA (A fragment reused across all 16 s-tiles) ----
    const bf16* qrow = Q + (size_t)(rb + l16) * DIM + h * HD;
    const v16bf a0 = ld_a_frag(qrow + half * 8);        // k = 0..31
    const v16bf a1 = ld_a_frag(qrow + 32 + half * 8);   // k = 32..63
    for (int j = 0; j < 16; ++j) {
      const int sbase = ((j << 3) + wave) << 4;         // wave w owns tiles w+8j
      const bf16* krow = Kp + (size_t)(sbase + l16) * DIM + h * HD + half * 8;
      const v16bf b0 = ld_a_frag(krow);
      const v16bf b1 = ld_a_frag(krow + 32);
      v8f acc = {};
      acc = __builtin_amdgcn_wmma_f32_16x16x32_bf16(false, a0, false, b0,
                                                    (short)0, acc, false, false);
      acc = __builtin_amdgcn_wmma_f32_16x16x32_bf16(false, a1, false, b1,
                                                    (short)0, acc, false, false);
      #pragma unroll
      for (int r = 0; r < 8; ++r) {
        const int m = r + half * 8;
        sc[m * 2048 + sbase + l16] = acc[r] * 0.125f;   // / sqrt(64)
      }
    }
    __syncthreads();

    // ---- row softmax over 2048, 16 threads per row ----
    float mx = -INFINITY;
    for (int c = sub; c < 2048; c += 16) mx = fmaxf(mx, sc[row * 2048 + c]);
    red[tid] = mx; __syncthreads();
    if (sub == 0) {
      float m2 = red[row * 16];
      #pragma unroll
      for (int i = 1; i < 16; ++i) m2 = fmaxf(m2, red[row * 16 + i]);
      rowm[row] = m2;
    }
    __syncthreads();
    const float M = rowm[row];
    float ls = 0.f;
    for (int c = sub; c < 2048; c += 16) ls += __expf(sc[row * 2048 + c] - M);
    red[tid] = ls; __syncthreads();
    if (sub == 0) {
      float s2 = 0.f;
      #pragma unroll
      for (int i = 0; i < 16; ++i) s2 += red[row * 16 + i];
      rowl[row] = s2;
    }
    __syncthreads();
    const float inv = 1.f / (16.f * rowl[row]);   // mean over heads folded in
    for (int c = sub; c < 2048; c += 16)
      accb[row * 2048 + c] += __expf(sc[row * 2048 + c] - M) * inv;
    __syncthreads();   // protect sc before next head overwrites it
  }

  for (int i = tid; i < 16 * 2048; i += 256)
    attn[(size_t)(rb + (i >> 11)) * 2048 + (i & 2047)] = accb[i];
}

// ---------------------------------------------------------------------------
// Kernel 3: copy gate log-probs: log(sigmoid(x)), log(1-sigmoid(x))
// ---------------------------------------------------------------------------
__global__ void copy_gate_kernel(const float* __restrict__ tgt, const float* __restrict__ wp,
                                 const float* __restrict__ bp,
                                 float* __restrict__ lcp, float* __restrict__ l1m) {
  const int wave = threadIdx.x >> 5, lane = threadIdx.x & 31;
  const int r = blockIdx.x * 8 + wave;
  const float* x = tgt + (size_t)r * DIM;
  float s = 0.f;
  for (int i = lane; i < DIM; i += 32) s += x[i] * wp[i];
  #pragma unroll
  for (int o = 16; o; o >>= 1) s += __shfl_xor(s, o, 32);
  if (lane == 0) {
    const float z = s + bp[0];
    lcp[r] = (z >= 0.f) ? -log1pf(expf(-z)) : (z - log1pf(expf(z)));   // log sig(z)
    l1m[r] = (z <= 0.f) ? -log1pf(expf(z))  : (-z - log1pf(expf(-z))); // log sig(-z)
  }
}

// ---------------------------------------------------------------------------
// Kernel 4: zero d_out (attn_vocab accumulator lives in d_out)
// ---------------------------------------------------------------------------
__global__ void zero_f32(float* __restrict__ p, int n) {
  const int i = blockIdx.x * 256 + threadIdx.x;
  if (i < n) p[i] = 0.f;
}

// ---------------------------------------------------------------------------
// Kernel 5: segment_sum scatter: out[t, ids[s]] += attn[t, s]
// ---------------------------------------------------------------------------
__global__ void scatter_kernel(const float* __restrict__ attn, const int* __restrict__ ids,
                               float* __restrict__ out) {
  const int i = blockIdx.x * 256 + threadIdx.x;   // 0 .. 2048*2048-1
  const int t = i >> 11, s = i & 2047;
  atomicAdd(out + (size_t)t * TOT + ids[s], attn[i]);
}

// ---------------------------------------------------------------------------
// Kernel 6: log_softmax(logits) + logaddexp combine, in place over d_out.
// One block per target row.
// ---------------------------------------------------------------------------
__global__ void finalize_kernel(const float* __restrict__ logits, const float* __restrict__ lcp,
                                const float* __restrict__ l1m, float* __restrict__ out) {
  __shared__ float sm[256], sl[256];
  const int t = blockIdx.x, tid = threadIdx.x;
  const float* row = logits + (size_t)t * VOC;
  // online max/sumexp
  float m = -INFINITY, l = 0.f;
  for (int v = tid; v < VOC; v += 256) {
    __builtin_prefetch(row + v + 4096, 0, 1);
    const float x = row[v];
    if (x > m) { l = l * __expf(m - x) + 1.f; m = x; }
    else       { l += __expf(x - m); }
  }
  sm[tid] = m; sl[tid] = l; __syncthreads();
  for (int off = 128; off; off >>= 1) {
    if (tid < off) {
      const float m2 = sm[tid + off], l2 = sl[tid + off];
      const float M = fmaxf(sm[tid], m2);
      sl[tid] = sl[tid] * __expf(sm[tid] - M) + l2 * __expf(m2 - M);
      sm[tid] = M;
    }
    __syncthreads();
  }
  const float logZ = sm[0] + __logf(sl[0]);
  const float a = l1m[t], b = lcp[t];
  float* orow = out + (size_t)t * TOT;
  const float LOG_EPS = -15.942385152878742f;   // log(FLT_EPSILON)
  for (int v = tid; v < TOT; v += 256) {
    const float av = orow[v];
    const float plp = ((av == 0.f) ? LOG_EPS : __logf(av)) + b;
    float r;
    if (v < VOC) {
      const float dlp = row[v] - logZ + a;
      const float mx = fmaxf(dlp, plp), mn = fminf(dlp, plp);
      r = mx + log1pf(__expf(mn - mx));
    } else {
      r = plp;   // decoder branch is -inf -> fmin; logsumexp degenerates to plp
    }
    orow[v] = r;
  }
}

// ---------------------------------------------------------------------------
extern "C" void kernel_launch(void* const* d_in, const int* in_sizes, int n_in,
                              void* d_out, int out_size, void* d_ws, size_t ws_size,
                              hipStream_t stream) {
  (void)in_sizes; (void)n_in; (void)out_size; (void)ws_size;
  const float* logits = (const float*)d_in[0];
  const int*   ids    = (const int*)  d_in[1];
  const float* src    = (const float*)d_in[2];
  const float* tgt    = (const float*)d_in[3];
  const float* w_q    = (const float*)d_in[4];
  const float* b_q    = (const float*)d_in[5];
  const float* w_k    = (const float*)d_in[6];
  const float* b_k    = (const float*)d_in[7];
  const float* w_ptr  = (const float*)d_in[8];
  const float* b_ptr  = (const float*)d_in[9];
  float* out = (float*)d_out;

  // workspace layout
  bf16*  Qb   = (bf16*)d_ws;                         // 2048*1024 bf16 (4 MB)
  bf16*  Kb   = Qb + (size_t)TGTN * DIM;             // 4 MB
  float* attn = (float*)(Kb + (size_t)SRCN * DIM);   // 2048*2048 f32 (16 MB)
  float* lcp  = attn + (size_t)TGTN * SRCN;          // 2048 f32
  float* l1m  = lcp + TGTN;                          // 2048 f32

  const dim3 gproj(TGTN / 16, DIM / 128);
  proj_gemm<<<gproj, 256, 0, stream>>>(tgt, w_q, b_q, Qb);
  proj_gemm<<<gproj, 256, 0, stream>>>(src, w_k, b_k, Kb);

  zero_f32<<<((TGTN * TOT) + 255) / 256, 256, 0, stream>>>(out, TGTN * TOT);
  copy_gate_kernel<<<TGTN / 8, 256, 0, stream>>>(tgt, w_ptr, b_ptr, lcp, l1m);

  const size_t ldsB = (size_t)(2 * 16 * 2048 + 256 + 32) * sizeof(float); // ~257 KB
  attn_kernel<<<TGTN / 16, 256, ldsB, stream>>>(Qb, Kb, attn);

  scatter_kernel<<<(TGTN * SRCN) / 256, 256, 0, stream>>>(attn, ids, out);
  finalize_kernel<<<TGTN, 256, 0, stream>>>(logits, lcp, l1m, out);
}
